// MLA_50457275794046
// MI455X (gfx1250) — matrix operand order repaired and checked by reading, not compiled, not run on hardware
//
#include <hip/hip_runtime.h>
#include <hip/hip_bf16.h>

// ---------------- problem constants ----------------
constexpr int S_TOK   = 2048;
constexpr int HIDDEN_ = 2048;
constexpr int NH      = 16;     // heads
constexpr int DN_     = 128;
constexpr int DR_     = 64;
constexpr int DV_     = 128;
constexpr int Q_LORA_ = 1536;
constexpr int KV_LORA_= 512;
constexpr int FUSED_N = Q_LORA_ + KV_LORA_ + DR_;     // 2112
constexpr int QDIM    = NH * (DN_ + DR_);             // 3072
constexpr int KVDIM   = NH * (DN_ + DV_);             // 4096
constexpr int ODIM    = NH * DV_;                     // 2048

typedef __attribute__((ext_vector_type(16))) __bf16 v16bf;
typedef __attribute__((ext_vector_type(8)))  __bf16 v8bf;
typedef __attribute__((ext_vector_type(8)))  float  v8f;

// Native bf16 convert (single v_cvt op on gfx1250, RNE default)
__device__ __forceinline__ unsigned short f2bf(float f) {
  union { __bf16 b; unsigned short u; } c;
  c.b = (__bf16)f;
  return c.u;
}

// Convert float4 -> 4 x bf16, single 8-byte LDS store (dst must be 8B aligned)
__device__ __forceinline__ void store_bf4(unsigned short* dst, float4 v) {
  union { unsigned short us[4]; uint2 u2; } p;
  p.us[0] = f2bf(v.x); p.us[1] = f2bf(v.y);
  p.us[2] = f2bf(v.z); p.us[3] = f2bf(v.w);
  *(uint2*)dst = p.u2;
}

__device__ __forceinline__ v8f wmma_bf16(v16bf a, v16bf b, v8f c) {
  return __builtin_amdgcn_wmma_f32_16x16x32_bf16(false, a, false, b, (short)0, c,
                                                 false, false);
}

// A fragment: lane m = l%16, chunks at k = h*8 and 16+h*8 (ISA 16-bit A layout)
__device__ __forceinline__ v16bf load_afrag(const unsigned short* row, int lh) {
  v16bf a;
  v8bf* ap = (v8bf*)&a;
  ap[0] = *(const v8bf*)(row + lh * 8);
  ap[1] = *(const v8bf*)(row + 16 + lh * 8);
  return a;
}
// B fragment: lane n = l%16, chunks at k = h*8 and 16+h*8 (matches A layout)
__device__ __forceinline__ v16bf load_bfrag(const unsigned short* row, int lh) {
  v16bf b;
  v8bf* bp = (v8bf*)&b;
  bp[0] = *(const v8bf*)(row + lh * 8);
  bp[1] = *(const v8bf*)(row + 16 + lh * 8);
  return b;
}

// ============================================================
// Generic GEMM: C[M,N] = A[M,K](lda) * B[K,N](ldb), f32 in/out,
// bf16 WMMA compute. Block 128x128x32, 256 threads (8 waves),
// double-buffered LDS, M assumed multiple of 128 (true at all
// call sites); NCHECK guards ragged N (GEMM1 only).
// ============================================================
template <bool NCHECK>
__global__ __launch_bounds__(256) void gemm_bf16_wmma(
    const float* __restrict__ A, int lda,
    const float* __restrict__ B, int ldb,
    float* __restrict__ C, int ldc,
    int M, int N, int K)
{
  constexpr int BKP = 40;                    // 32 + 8 pad, rows stay 16B aligned
  __shared__ unsigned short sA[2][128][BKP]; // row-major M x K tile
  __shared__ unsigned short sB[2][128][BKP]; // transposed: [n][k]

  const int tid  = threadIdx.x;
  const int wave = tid >> 5;
  const int lane = tid & 31;
  const int l16  = lane & 15;
  const int lh   = lane >> 4;
  const int wm   = wave >> 1;                // 0..3 along M
  const int wn   = wave & 1;                 // 0..1 along N
  const int bm   = blockIdx.y * 128;
  const int bn   = blockIdx.x * 128;

  const v8f vzero = {0.f,0.f,0.f,0.f,0.f,0.f,0.f,0.f};
  v8f acc[2][4];
#pragma unroll
  for (int i = 0; i < 2; ++i)
#pragma unroll
    for (int j = 0; j < 4; ++j) acc[i][j] = vzero;

  auto stage = [&](int k0, int db) {
    // ---- A tile (128x32): packed b64 LDS stores ----
#pragma unroll
    for (int p = 0; p < 4; ++p) {
      int idx = p * 256 + tid;
      int r = idx >> 3;
      int c = (idx & 7) << 2;
      float4 v = *(const float4*)(A + (size_t)(bm + r) * lda + k0 + c);
      store_bf4(&sA[db][r][c], v);
    }
    // ---- B tile (32x128 -> [n][k] transpose), scalar b16 stores ----
#pragma unroll
    for (int p = 0; p < 4; ++p) {
      int idx = p * 256 + tid;
      int kk = idx >> 5;
      int n4 = (idx & 31) << 2;
      const float* src = B + (size_t)(k0 + kk) * ldb + bn + n4;
      float4 v = make_float4(0.f, 0.f, 0.f, 0.f);
      if (!NCHECK || (bn + n4 + 3 < N)) {
        v = *(const float4*)src;
      } else {
        if (bn + n4 + 0 < N) v.x = src[0];
        if (bn + n4 + 1 < N) v.y = src[1];
        if (bn + n4 + 2 < N) v.z = src[2];
        if (bn + n4 + 3 < N) v.w = src[3];
      }
      sB[db][n4 + 0][kk] = f2bf(v.x); sB[db][n4 + 1][kk] = f2bf(v.y);
      sB[db][n4 + 2][kk] = f2bf(v.z); sB[db][n4 + 3][kk] = f2bf(v.w);
    }
  };

  stage(0, 0);
  __syncthreads();

  const int nk = K >> 5;
  for (int kt = 0; kt < nk; ++kt) {
    const int db = kt & 1;
    if (kt + 1 < nk) {
      stage((kt + 1) << 5, db ^ 1);                     // overlap with compute
      if (kt + 2 < nk) {                                // L2 prefetch of K+2 tile
        __builtin_prefetch(A + (size_t)(bm + (tid >> 3)) * lda + ((kt + 2) << 5), 0, 3);
        __builtin_prefetch(B + (size_t)(((kt + 2) << 5) + (tid >> 5)) * ldb + bn, 0, 3);
      }
    }
    // ---- hoist all fragments, then burst 8 WMMAs ----
    v16bf af[2], bf[4];
#pragma unroll
    for (int i = 0; i < 2; ++i)
      af[i] = load_afrag(&sA[db][wm * 32 + i * 16 + l16][0], lh);
#pragma unroll
    for (int j = 0; j < 4; ++j)
      bf[j] = load_bfrag(&sB[db][wn * 64 + j * 16 + l16][0], lh);
#pragma unroll
    for (int j = 0; j < 4; ++j)
#pragma unroll
      for (int i = 0; i < 2; ++i)
        acc[i][j] = wmma_bf16(af[i], bf[j], acc[i][j]);
    __syncthreads();
  }

  // ---- epilogue: row = r + 8*half, col = lane%16 ----
#pragma unroll
  for (int i = 0; i < 2; ++i)
#pragma unroll
    for (int j = 0; j < 4; ++j)
#pragma unroll
      for (int r = 0; r < 8; ++r) {
        int row = bm + wm * 32 + i * 16 + 8 * lh + r;
        int col = bn + wn * 64 + j * 16 + l16;
        if (!NCHECK || col < N) C[(size_t)row * ldc + col] = acc[i][j][r];
      }
}

// ============================================================
// RMS-norm over a column slice of a strided buffer (in place)
// ============================================================
__global__ __launch_bounds__(256) void rmsnorm_kernel(
    float* __restrict__ buf, int stride, int coff, int D,
    const float* __restrict__ w)
{
  __shared__ float red[8];
  float* x = buf + (size_t)blockIdx.x * stride + coff;
  float ss = 0.f;
  for (int i = threadIdx.x; i < D; i += 256) { float v = x[i]; ss += v * v; }
#pragma unroll
  for (int m = 16; m >= 1; m >>= 1) ss += __shfl_xor(ss, m, 32);
  if ((threadIdx.x & 31) == 0) red[threadIdx.x >> 5] = ss;
  __syncthreads();
  if (threadIdx.x == 0) {
    float t = 0.f;
#pragma unroll
    for (int i = 0; i < 8; ++i) t += red[i];
    red[0] = __frsqrt_rn(t / (float)D + 1e-6f);
  }
  __syncthreads();
  float s = red[0];
  for (int i = threadIdx.x; i < D; i += 256) x[i] = x[i] * s * w[i];
}

// ============================================================
// RoPE (dim 64, half-split form as in reference)
// ============================================================
__device__ __forceinline__ void rope_apply(float* base, int pos, int j) {
  float inv = __powf(10000.0f, -(float)j / 32.0f);
  float f = (float)pos * inv;
  float sn, cs;
  __sincosf(f, &sn, &cs);
  float x1 = base[j], x2 = base[j + 32];
  base[j]      = x1 * cs - x2 * sn;
  base[j + 32] = x2 * cs + x1 * sn;
}

__global__ void rope_k_kernel(float* __restrict__ kpe, int stride,
                              const int* __restrict__ pos) {
  rope_apply(kpe + (size_t)blockIdx.x * stride, pos[blockIdx.x], threadIdx.x);
}

__global__ void rope_q_kernel(float* __restrict__ q, const int* __restrict__ pos) {
  float* base = q + (size_t)blockIdx.x * QDIM + blockIdx.y * (DN_ + DR_) + DN_;
  rope_apply(base, pos[blockIdx.x], threadIdx.x);
}

// ============================================================
// Flash attention, one (head, 64-row q-block) per workgroup.
// 4 waves; wave w owns q rows [qb+16w, qb+16w+16) x all keys.
// ============================================================
__global__ __launch_bounds__(128) void mla_attn_kernel(
    const float* __restrict__ q,    // (S, 3072): per head 128 nope + 64 rope
    const float* __restrict__ kv,   // (S, 4096): per head 128 k_nope + 128 v
    const float* __restrict__ kpe,  // (S, stride 2112): 64 roped, shared by heads
    float* __restrict__ o)          // (S, 2048)
{
  constexpr int DH = DN_ + DR_;     // 192
  constexpr int QP = 200;           // padded row (ushort), 400B = 25*16
  constexpr int VP = 72;            // padded row, 144B = 9*16
  __shared__ unsigned short sQ[64][QP];
  __shared__ unsigned short sK[64][QP];
  __shared__ unsigned short sVT[DV_][VP];   // V transposed: [d][key]
  __shared__ unsigned short sP[4][16][VP];  // per-wave P tile (16x64)

  const int tid  = threadIdx.x;
  const int wave = tid >> 5;
  const int lane = tid & 31;
  const int l16  = lane & 15;
  const int lh   = lane >> 4;
  const int h    = blockIdx.y;
  const int qb   = blockIdx.x * 64;
  const float scale = 0.07216878364870323f;   // 1/sqrt(192)

  // ---- stage Q tile (64 x 192) once ----
#pragma unroll
  for (int p = 0; p < 24; ++p) {
    int idx = p * 128 + tid;
    int r = idx / 48;
    int c4 = (idx % 48) * 4;
    float4 v = *(const float4*)(q + (size_t)(qb + r) * QDIM + h * DH + c4);
    store_bf4(&sQ[r][c4], v);
  }

  const v8f vzero = {0.f,0.f,0.f,0.f,0.f,0.f,0.f,0.f};
  v8f oacc[8];
#pragma unroll
  for (int t = 0; t < 8; ++t) oacc[t] = vzero;
  float m_prev[8], l_prev[8];
#pragma unroll
  for (int r = 0; r < 8; ++r) { m_prev[r] = -1e30f; l_prev[r] = 0.f; }

  const int nkb = qb / 64 + 1;                // causal block bound
  for (int kb = 0; kb < nkb; ++kb) {
    const int k0 = kb * 64;
    // ---- stage K tile (64 x 192): k_nope ++ broadcast k_pe ----
#pragma unroll
    for (int p = 0; p < 24; ++p) {
      int idx = p * 128 + tid;
      int r = idx / 48;
      int c4 = (idx % 48) * 4;
      float4 v;
      if (c4 < DN_)
        v = *(const float4*)(kv + (size_t)(k0 + r) * KVDIM + h * (DN_ + DV_) + c4);
      else
        v = *(const float4*)(kpe + (size_t)(k0 + r) * FUSED_N + (c4 - DN_));
      store_bf4(&sK[r][c4], v);
    }
    // ---- stage V transposed (128 x 64) ----
#pragma unroll
    for (int p = 0; p < 16; ++p) {
      int idx = p * 128 + tid;
      int r = idx / 32;                 // key token
      int d4 = (idx % 32) * 4;
      float4 v = *(const float4*)(kv + (size_t)(k0 + r) * KVDIM +
                                  h * (DN_ + DV_) + DN_ + d4);
      sVT[d4 + 0][r] = f2bf(v.x); sVT[d4 + 1][r] = f2bf(v.y);
      sVT[d4 + 2][r] = f2bf(v.z); sVT[d4 + 3][r] = f2bf(v.w);
    }
    __syncthreads();

    // ---- S = Q K^T  (16 x 64 per wave; hoist frags, burst WMMAs) ----
    v8f sacc[4];
#pragma unroll
    for (int j = 0; j < 4; ++j) sacc[j] = vzero;
#pragma unroll
    for (int ks = 0; ks < 6; ++ks) {
      v16bf afr = load_afrag(&sQ[wave * 16 + l16][ks * 32], lh);
      v16bf bfr[4];
#pragma unroll
      for (int j = 0; j < 4; ++j)
        bfr[j] = load_bfrag(&sK[j * 16 + l16][ks * 32], lh);
#pragma unroll
      for (int j = 0; j < 4; ++j)
        sacc[j] = wmma_bf16(afr, bfr[j], sacc[j]);
    }

    // ---- online softmax update (wave owns full rows) ----
    float sf[8];
    const int row0 = qb + wave * 16 + 8 * lh;
#pragma unroll
    for (int r = 0; r < 8; ++r) {
      const int grow = row0 + r;
      float pv[4];
      float mx = -1e30f;
#pragma unroll
      for (int j = 0; j < 4; ++j) {
        int gcol = k0 + j * 16 + l16;
        float s = sacc[j][r] * scale;
        if (gcol > grow) s = -1e30f;          // causal mask
        pv[j] = s;
        mx = fmaxf(mx, s);
      }
      mx = fmaxf(mx, __shfl_xor(mx, 1, 32));
      mx = fmaxf(mx, __shfl_xor(mx, 2, 32));
      mx = fmaxf(mx, __shfl_xor(mx, 4, 32));
      mx = fmaxf(mx, __shfl_xor(mx, 8, 32));
      float mn = fmaxf(m_prev[r], mx);
      float f  = __expf(m_prev[r] - mn);
      float rs = 0.f;
#pragma unroll
      for (int j = 0; j < 4; ++j) {
        float pe = __expf(pv[j] - mn);
        rs += pe;
        sP[wave][8 * lh + r][j * 16 + l16] = f2bf(pe);
      }
      rs += __shfl_xor(rs, 1, 32);
      rs += __shfl_xor(rs, 2, 32);
      rs += __shfl_xor(rs, 4, 32);
      rs += __shfl_xor(rs, 8, 32);
      l_prev[r] = l_prev[r] * f + rs;
      m_prev[r] = mn;
      sf[r] = f;
    }
#pragma unroll
    for (int t = 0; t < 8; ++t)
#pragma unroll
      for (int r = 0; r < 8; ++r) oacc[t][r] *= sf[r];

    // ---- O += P V  (per-wave LDS round-trip; same-wave DS ops in order) ----
#pragma unroll
    for (int ks = 0; ks < 2; ++ks) {
      v16bf pfr = load_afrag(&sP[wave][l16][ks * 32], lh);
      v16bf vfr[8];
#pragma unroll
      for (int t = 0; t < 8; ++t)
        vfr[t] = load_bfrag(&sVT[t * 16 + l16][ks * 32], lh);
#pragma unroll
      for (int t = 0; t < 8; ++t)
        oacc[t] = wmma_bf16(pfr, vfr[t], oacc[t]);
    }
    __syncthreads();
  }

  // ---- normalize + write ----
#pragma unroll
  for (int r = 0; r < 8; ++r) {
    float inv = 1.0f / l_prev[r];
    int grow = qb + wave * 16 + 8 * lh + r;
#pragma unroll
    for (int t = 0; t < 8; ++t)
      o[(size_t)grow * ODIM + h * DV_ + t * 16 + l16] = oacc[t][r] * inv;
  }
}

// ============================================================
extern "C" void kernel_launch(void* const* d_in, const int* in_sizes, int n_in,
                              void* d_out, int out_size, void* d_ws, size_t ws_size,
                              hipStream_t stream) {
  (void)in_sizes; (void)n_in; (void)out_size; (void)ws_size;
  const float* hidden    = (const float*)d_in[0];
  const int*   pos       = (const int*)d_in[1];
  const float* w_fused_a = (const float*)d_in[2];
  const float* q_ln      = (const float*)d_in[3];
  const float* kv_ln     = (const float*)d_in[4];
  const float* w_q_b     = (const float*)d_in[5];
  const float* w_kv_b    = (const float*)d_in[6];
  const float* w_o       = (const float*)d_in[7];
  float* out = (float*)d_out;

  float* fused = (float*)d_ws;                       // S x 2112
  float* qbuf  = fused + (size_t)S_TOK * FUSED_N;    // S x 3072
  float* kvbuf = qbuf  + (size_t)S_TOK * QDIM;       // S x 4096
  float* attn  = kvbuf + (size_t)S_TOK * KVDIM;      // S x 2048

  dim3 b256(256);
  // 1) fused = hidden @ w_fused_a  (N=2112 ragged -> NCHECK)
  gemm_bf16_wmma<true><<<dim3((FUSED_N + 127) / 128, S_TOK / 128), b256, 0, stream>>>(
      hidden, HIDDEN_, w_fused_a, FUSED_N, fused, FUSED_N, S_TOK, FUSED_N, HIDDEN_);
  // 2) RMS norms (in place, column slices of fused)
  rmsnorm_kernel<<<S_TOK, 256, 0, stream>>>(fused, FUSED_N, 0, Q_LORA_, q_ln);
  rmsnorm_kernel<<<S_TOK, 256, 0, stream>>>(fused, FUSED_N, Q_LORA_, KV_LORA_, kv_ln);
  // 3) RoPE on k_pe slice (cols 2048..2111 of fused)
  rope_k_kernel<<<S_TOK, 32, 0, stream>>>(fused + Q_LORA_ + KV_LORA_, FUSED_N, pos);
  // 4) q = q_a @ w_q_b
  gemm_bf16_wmma<false><<<dim3(QDIM / 128, S_TOK / 128), b256, 0, stream>>>(
      fused, FUSED_N, w_q_b, QDIM, qbuf, QDIM, S_TOK, QDIM, Q_LORA_);
  // 5) RoPE on q_pe per head
  rope_q_kernel<<<dim3(S_TOK, NH), 32, 0, stream>>>(qbuf, pos);
  // 6) kv = kv_a @ w_kv_b
  gemm_bf16_wmma<false><<<dim3(KVDIM / 128, S_TOK / 128), b256, 0, stream>>>(
      fused + Q_LORA_, FUSED_N, w_kv_b, KVDIM, kvbuf, KVDIM, S_TOK, KVDIM, KV_LORA_);
  // 7) flash attention -> attn (S, 2048)
  mla_attn_kernel<<<dim3(S_TOK / 64, NH), 128, 0, stream>>>(
      qbuf, kvbuf, fused + Q_LORA_ + KV_LORA_, attn);
  // 8) out = attn @ w_o
  gemm_bf16_wmma<false><<<dim3(HIDDEN_ / 128, S_TOK / 128), b256, 0, stream>>>(
      attn, ODIM, w_o, HIDDEN_, out, HIDDEN_, S_TOK, HIDDEN_, ODIM);
}
